// AlignerModel_25400436588694
// MI455X (gfx1250) — matrix operand order repaired and checked by reading, not compiled, not run on hardware
//
#include <hip/hip_runtime.h>
#include <hip/hip_bf16.h>

#define B_SZ   32
#define T1_SZ  2048
#define T2_SZ  512
#define CMEL   80
#define CTXT   512
#define CATT   128
#define TEMP   0.0005f

typedef __attribute__((ext_vector_type(16))) _Float16 v16h;
typedef __attribute__((ext_vector_type(8)))  _Float16 v8h;
typedef __attribute__((ext_vector_type(8)))  float    v8f;

// A-fragment (16-bit A 16x32): halves 0..15 = K(ko+16h..+15),
// contiguous, matching B. rowp points at A row + ko.
__device__ __forceinline__ v16h frag_a(const _Float16* rowp, int hh) {
    v8h lo = *(const v8h*)(rowp + 16 * hh);
    v8h hi = *(const v8h*)(rowp + 16 * hh + 8);
    return __builtin_shufflevector(lo, hi,
        0,1,2,3,4,5,6,7,8,9,10,11,12,13,14,15);
}
__device__ __forceinline__ v8f wmma_(v16h a, v16h b, v8f c) {
    return __builtin_amdgcn_wmma_f32_16x16x32_f16(false, a, false, b,
                                                  (short)0, c, false, false);
}

// ---------------------------------------------------------------------------
// Prep: cast inputs to f16 with zero pad rows / padded channels.
// textf: (B, T2+2, 512), row t+1 (rows 0 and T2+1 are zero)
// specf: (B, T1+4, 96),  row t+2 (2 zero rows each end; ci 80..95 zero)
// ---------------------------------------------------------------------------
__global__ void convert_text_kernel(const float* __restrict__ text,
                                    _Float16* __restrict__ textf) {
    size_t i = (size_t)blockIdx.x * blockDim.x + threadIdx.x;
    size_t total = (size_t)B_SZ * (T2_SZ + 2) * CTXT;
    if (i >= total) return;
    int c = (int)(i % CTXT);
    size_t rowi = i / CTXT;
    int row = (int)(rowi % (T2_SZ + 2));
    int b = (int)(rowi / (T2_SZ + 2));
    int t = row - 1;
    textf[i] = (t >= 0 && t < T2_SZ)
             ? (_Float16)text[((size_t)b * T2_SZ + t) * CTXT + c]
             : (_Float16)0.f;
}

__global__ void convert_spec_kernel(const float* __restrict__ spec,
                                    _Float16* __restrict__ specf) {
    size_t i = (size_t)blockIdx.x * blockDim.x + threadIdx.x;
    size_t total = (size_t)B_SZ * (T1_SZ + 4) * 96;
    if (i >= total) return;
    int c = (int)(i % 96);
    size_t rowi = i / 96;
    int row = (int)(rowi % (T1_SZ + 4));
    int b = (int)(rowi / (T1_SZ + 4));
    int t = row - 2;
    specf[i] = (t >= 0 && t < T1_SZ && c < CMEL)
             ? (_Float16)spec[((size_t)b * T1_SZ + t) * CMEL + c]
             : (_Float16)0.f;
}

// Repack conv weights to f16 [tap][co][ci] (A-matrix layout, zero padded ci).
#define NPK1 (3 * CATT * CTXT)     // kw1
#define NPK2 (CATT * CATT)         // kw2
#define NPQ1 (3 * 160 * 96)        // qw1 (ci 80->96)
#define NPQ2 (3 * 80 * 160)        // qw2
#define NPQ3 (CATT * 96)           // qw3 (ci 80->96)
__global__ void pack_weights_kernel(
    const float* __restrict__ kw1, const float* __restrict__ kw2,
    const float* __restrict__ qw1, const float* __restrict__ qw2,
    const float* __restrict__ qw3,
    _Float16* __restrict__ pkw1, _Float16* __restrict__ pkw2,
    _Float16* __restrict__ pqw1, _Float16* __restrict__ pqw2,
    _Float16* __restrict__ pqw3) {
    int i = blockIdx.x * blockDim.x + threadIdx.x;
    if (i < NPK1) {
        int tap = i / (CATT * CTXT); int r = i % (CATT * CTXT);
        int co = r / CTXT, ci = r % CTXT;
        pkw1[i] = (_Float16)kw1[((size_t)co * CTXT + ci) * 3 + tap];
        return;
    }
    i -= NPK1;
    if (i < NPK2) { pkw2[i] = (_Float16)kw2[i]; return; }
    i -= NPK2;
    if (i < NPQ1) {
        int tap = i / (160 * 96); int r = i % (160 * 96);
        int co = r / 96, ci = r % 96;
        pqw1[i] = (_Float16)((ci < CMEL)
                    ? qw1[((size_t)co * CMEL + ci) * 3 + tap] : 0.f);
        return;
    }
    i -= NPQ1;
    if (i < NPQ2) {
        int tap = i / (80 * 160); int r = i % (80 * 160);
        int co = r / 160, ci = r % 160;
        pqw2[i] = (_Float16)qw2[((size_t)co * 160 + ci) * 3 + tap];
        return;
    }
    i -= NPQ2;
    if (i < NPQ3) {
        int co = i / 96, ci = i % 96;
        pqw3[i] = (_Float16)((ci < CMEL) ? qw3[(size_t)co * CMEL + ci] : 0.f);
        return;
    }
}

// ---------------------------------------------------------------------------
// K encoder, WMMA: conv(512->128,k3)+relu as 3 shifted GEMMs, then 1x1 GEMM.
// Block = 8 waves, 64 positions. Wave w owns co-tile w (M), 4 N-tiles.
// B-fragments batched ahead of the WMMA group for partial s_wait overlap.
// ---------------------------------------------------------------------------
__global__ __launch_bounds__(256) void k_enc_wmma(
    const _Float16* __restrict__ textf,
    const _Float16* __restrict__ pkw1, const float* __restrict__ kb1,
    const _Float16* __restrict__ pkw2, const float* __restrict__ kb2,
    _Float16* __restrict__ kf, float* __restrict__ k2)
{
    __shared__ _Float16 h1[64][CATT];
    __shared__ float k2s[64];
    const int b  = blockIdx.y;
    const int n0 = blockIdx.x * 64;
    const int tid = threadIdx.x;
    const int w = tid >> 5, lane = tid & 31, l16 = lane & 15, hh = lane >> 4;

    for (int i = tid; i < 64; i += 256) k2s[i] = 0.f;

    v8f zero = {0,0,0,0,0,0,0,0};
    v8f acc[4] = {zero, zero, zero, zero};
    // conv1: y[co][n] += sum_tap W_tap[co][ci] * x[n+tap-1][ci]
    for (int tap = 0; tap < 3; ++tap) {
        const _Float16* abase = pkw1 + ((size_t)tap * CATT + w * 16 + l16) * CTXT;
        const _Float16* bb[4];
        #pragma unroll
        for (int j = 0; j < 4; ++j)   // textf row = (n+tap-1) + 1 = n + tap
            bb[j] = textf + ((size_t)b * (T2_SZ + 2) + n0 + j * 16 + l16 + tap) * CTXT
                  + 16 * hh;
        for (int ko = 0; ko < CTXT; ko += 32) {
            v16h a = frag_a(abase + ko, hh);
            v16h bf[4];
            #pragma unroll
            for (int j = 0; j < 4; ++j) bf[j] = *(const v16h*)(bb[j] + ko);
            #pragma unroll
            for (int j = 0; j < 4; ++j) acc[j] = wmma_(a, bf[j], acc[j]);
        }
    }
    {   // bias + relu -> LDS h1[t][co] (B-frag layout for conv2)
        float bias[8];
        #pragma unroll
        for (int r = 0; r < 8; ++r) bias[r] = kb1[w * 16 + 8 * hh + r];
        #pragma unroll
        for (int j = 0; j < 4; ++j) {
            v8h hv;
            #pragma unroll
            for (int r = 0; r < 8; ++r)
                hv[r] = (_Float16)fmaxf(acc[j][r] + bias[r], 0.f);
            *(v8h*)&h1[j * 16 + l16][w * 16 + 8 * hh] = hv;
        }
    }
    __syncthreads();

    {   // conv2 1x1: K = 128
        v8f acc2[4] = {zero, zero, zero, zero};
        const _Float16* abase = pkw2 + (size_t)(w * 16 + l16) * CATT;
        #pragma unroll
        for (int ko = 0; ko < CATT; ko += 32) {
            v16h a = frag_a(abase + ko, hh);
            v16h bf[4];
            #pragma unroll
            for (int j = 0; j < 4; ++j)
                bf[j] = *(const v16h*)(&h1[j * 16 + l16][ko + 16 * hh]);
            #pragma unroll
            for (int j = 0; j < 4; ++j) acc2[j] = wmma_(a, bf[j], acc2[j]);
        }
        float bias[8];
        #pragma unroll
        for (int r = 0; r < 8; ++r) bias[r] = kb2[w * 16 + 8 * hh + r];
        #pragma unroll
        for (int j = 0; j < 4; ++j) {
            int t = n0 + j * 16 + l16;
            v8h hv; float ss = 0.f;
            #pragma unroll
            for (int r = 0; r < 8; ++r) {
                float v = acc2[j][r] + bias[r];
                hv[r] = (_Float16)v; ss += v * v;
            }
            *(v8h*)&kf[((size_t)b * T2_SZ + t) * CATT + w * 16 + 8 * hh] = hv;
            atomicAdd(&k2s[j * 16 + l16], ss);
        }
    }
    __syncthreads();
    for (int i = tid; i < 64; i += 256)
        k2[(size_t)b * T2_SZ + n0 + i] = k2s[i];
}

// ---------------------------------------------------------------------------
// Q encoder, WMMA: conv(80->160,k3)+relu -> conv(160->80,k3)+relu ->
// conv(80->128,1x1). 10 waves, 64 positions (+/-8 halo for conv2's +/-1).
// ---------------------------------------------------------------------------
__global__ __launch_bounds__(320) void q_enc_wmma(
    const _Float16* __restrict__ specf,
    const _Float16* __restrict__ pqw1, const float* __restrict__ qb1,
    const _Float16* __restrict__ pqw2, const float* __restrict__ qb2,
    const _Float16* __restrict__ pqw3, const float* __restrict__ qb3,
    _Float16* __restrict__ qf, float* __restrict__ q2)
{
    __shared__ _Float16 h1[80][160];   // positions n0-8 .. n0+71
    __shared__ _Float16 h2[64][96];    // ci padded 80->96
    __shared__ float q2s[64];
    const int b  = blockIdx.y;
    const int n0 = blockIdx.x * 64;
    const int tid = threadIdx.x;
    const int w = tid >> 5, lane = tid & 31, l16 = lane & 15, hh = lane >> 4;

    for (int i = tid; i < 64; i += 320) q2s[i] = 0.f;
    for (int i = tid; i < 64 * 16; i += 320)      // zero h2 pad columns
        h2[i >> 4][CMEL + (i & 15)] = (_Float16)0.f;

    v8f zero = {0,0,0,0,0,0,0,0};
    {   // conv1: M=160 (wave w = co tile), N = 5 tiles (pos n0-8 .. n0+71), K=96x3
        v8f acc[5] = {zero, zero, zero, zero, zero};
        for (int tap = 0; tap < 3; ++tap) {
            const _Float16* abase = pqw1 + ((size_t)tap * 160 + w * 16 + l16) * 96;
            const _Float16* bb[5];
            #pragma unroll
            for (int j = 0; j < 5; ++j) {
                int p = n0 - 8 + j * 16 + l16;      // output position
                int row = p + tap - 1 + 2;          // specf row (t+2)
                row = row < 0 ? 0 : (row > T1_SZ + 3 ? T1_SZ + 3 : row); // clamp:
                bb[j] = specf + ((size_t)b * (T1_SZ + 4) + row) * 96 + 16 * hh;
            }   // clamped rows only feed halo positions that get zeroed below
            #pragma unroll
            for (int ko = 0; ko < 96; ko += 32) {
                v16h a = frag_a(abase + ko, hh);
                v16h bf[5];
                #pragma unroll
                for (int j = 0; j < 5; ++j) bf[j] = *(const v16h*)(bb[j] + ko);
                #pragma unroll
                for (int j = 0; j < 5; ++j) acc[j] = wmma_(a, bf[j], acc[j]);
            }
        }
        float bias[8];
        #pragma unroll
        for (int r = 0; r < 8; ++r) bias[r] = qb1[w * 16 + 8 * hh + r];
        #pragma unroll
        for (int j = 0; j < 5; ++j) {
            int p = n0 - 8 + j * 16 + l16;
            bool valid = (p >= 0 && p < T1_SZ);     // conv2 input zero-padding
            v8h hv;
            #pragma unroll
            for (int r = 0; r < 8; ++r)
                hv[r] = (_Float16)(valid ? fmaxf(acc[j][r] + bias[r], 0.f) : 0.f);
            *(v8h*)&h1[j * 16 + l16][w * 16 + 8 * hh] = hv;
        }
    }
    __syncthreads();

    {   // conv2: M=80 (5 tiles) x N=64 (4 tiles) = 20 tiles over 10 waves
        int mt = w % 5;
        int jb = (w / 5) * 2;
        v8f acc2[2] = {zero, zero};
        for (int tap = 0; tap < 3; ++tap) {
            const _Float16* abase = pqw2 + ((size_t)tap * 80 + mt * 16 + l16) * 160;
            #pragma unroll
            for (int ko = 0; ko < 160; ko += 32) {
                v16h a = frag_a(abase + ko, hh);
                v16h bf[2];
                #pragma unroll
                for (int jj = 0; jj < 2; ++jj) {
                    int row = (jb + jj) * 16 + l16 + tap + 7;  // p_in - (n0-8)
                    bf[jj] = *(const v16h*)(&h1[row][ko + 16 * hh]);
                }
                #pragma unroll
                for (int jj = 0; jj < 2; ++jj) acc2[jj] = wmma_(a, bf[jj], acc2[jj]);
            }
        }
        float bias[8];
        #pragma unroll
        for (int r = 0; r < 8; ++r) bias[r] = qb2[mt * 16 + 8 * hh + r];
        #pragma unroll
        for (int jj = 0; jj < 2; ++jj) {
            v8h hv;
            #pragma unroll
            for (int r = 0; r < 8; ++r)
                hv[r] = (_Float16)fmaxf(acc2[jj][r] + bias[r], 0.f);
            *(v8h*)&h2[(jb + jj) * 16 + l16][mt * 16 + 8 * hh] = hv;
        }
    }
    __syncthreads();

    if (w < 8) {   // conv3 1x1: M=128 (8 tiles) x N=64 (4 tiles), K=96
        v8f acc3[4] = {zero, zero, zero, zero};
        const _Float16* abase = pqw3 + (size_t)(w * 16 + l16) * 96;
        #pragma unroll
        for (int ko = 0; ko < 96; ko += 32) {
            v16h a = frag_a(abase + ko, hh);
            v16h bf[4];
            #pragma unroll
            for (int j = 0; j < 4; ++j)
                bf[j] = *(const v16h*)(&h2[j * 16 + l16][ko + 16 * hh]);
            #pragma unroll
            for (int j = 0; j < 4; ++j) acc3[j] = wmma_(a, bf[j], acc3[j]);
        }
        float bias[8];
        #pragma unroll
        for (int r = 0; r < 8; ++r) bias[r] = qb3[w * 16 + 8 * hh + r];
        #pragma unroll
        for (int j = 0; j < 4; ++j) {
            int t = n0 + j * 16 + l16;
            v8h hv; float ss = 0.f;
            #pragma unroll
            for (int r = 0; r < 8; ++r) {
                float v = acc3[j][r] + bias[r];
                hv[r] = (_Float16)v; ss += v * v;
            }
            *(v8h*)&qf[((size_t)b * T1_SZ + t) * CATT + w * 16 + 8 * hh] = hv;
            atomicAdd(&q2s[j * 16 + l16], ss);
        }
    }
    __syncthreads();
    for (int i = tid; i < 64; i += 320)
        q2[(size_t)b * T1_SZ + n0 + i] = q2s[i];
}

// ---------------------------------------------------------------------------
// Fused qk GEMM (WMMA) + dist + mask + log-softmax over T2.
// ---------------------------------------------------------------------------
__global__ __launch_bounds__(256) void attn_kernel(
    const _Float16* __restrict__ qf, const _Float16* __restrict__ kf,
    const float* __restrict__ q2, const float* __restrict__ k2,
    const unsigned char* __restrict__ mask,
    float* __restrict__ out)
{
    const long long TOT = (long long)B_SZ * T1_SZ * T2_SZ;
    const int b    = blockIdx.y;
    const int m0   = blockIdx.x * 16;
    const int tid  = threadIdx.x;
    const int w    = tid >> 5;
    const int lane = tid & 31;
    const int l16  = lane & 15;
    const int hh   = lane >> 4;
    const int n0   = w * 64;

    __shared__ float red[8][16];
    __shared__ float rowmax_s[16];
    __shared__ float rowlse_s[16];

    v8f zero = {0,0,0,0,0,0,0,0};
    v8f acc[4] = {zero, zero, zero, zero};

    const _Float16* qbase = qf + ((size_t)b * T1_SZ + m0 + l16) * CATT;
    const _Float16* kbase = kf + ((size_t)b * T2_SZ + n0 + l16) * CATT + 16 * hh;

    #pragma unroll
    for (int ko = 0; ko < CATT; ko += 32) {
        v16h a = frag_a(qbase + ko, hh);
        v16h bf[4];
        #pragma unroll
        for (int j = 0; j < 4; ++j)
            bf[j] = *(const v16h*)(kbase + (size_t)j * 16 * CATT + ko);
        #pragma unroll
        for (int j = 0; j < 4; ++j) acc[j] = wmma_(a, bf[j], acc[j]);
    }

    float q2v[8];
    #pragma unroll
    for (int r = 0; r < 8; ++r)
        q2v[r] = q2[(size_t)b * T1_SZ + m0 + 8 * hh + r];

    float s[4][8];
    #pragma unroll
    for (int j = 0; j < 4; ++j) {
        int n = n0 + j * 16 + l16;
        float k2v = k2[(size_t)b * T2_SZ + n];
        bool  mk  = mask[(size_t)b * T2_SZ + n] != 0;
        #pragma unroll
        for (int r = 0; r < 8; ++r) {
            float d = q2v[r] + k2v - 2.0f * acc[j][r];
            float v = -TEMP * d;
            s[j][r] = mk ? -__builtin_inff() : v;
        }
    }

    float pmax[8];
    #pragma unroll
    for (int r = 0; r < 8; ++r) {
        float v = fmaxf(fmaxf(s[0][r], s[1][r]), fmaxf(s[2][r], s[3][r]));
        v = fmaxf(v, __shfl_xor(v, 1));
        v = fmaxf(v, __shfl_xor(v, 2));
        v = fmaxf(v, __shfl_xor(v, 4));
        v = fmaxf(v, __shfl_xor(v, 8));
        pmax[r] = v;
    }
    if (l16 == 0) {
        #pragma unroll
        for (int r = 0; r < 8; ++r) red[w][8 * hh + r] = pmax[r];
    }
    __syncthreads();
    if (tid < 16) {
        float m = red[0][tid];
        #pragma unroll
        for (int ww = 1; ww < 8; ++ww) m = fmaxf(m, red[ww][tid]);
        rowmax_s[tid] = m;
    }
    __syncthreads();

    float psum[8];
    #pragma unroll
    for (int r = 0; r < 8; ++r) {
        float rm = rowmax_s[8 * hh + r];
        float t = 0.0f;
        #pragma unroll
        for (int j = 0; j < 4; ++j) t += expf(s[j][r] - rm);
        t += __shfl_xor(t, 1);
        t += __shfl_xor(t, 2);
        t += __shfl_xor(t, 4);
        t += __shfl_xor(t, 8);
        psum[r] = t;
    }
    if (l16 == 0) {
        #pragma unroll
        for (int r = 0; r < 8; ++r) red[w][8 * hh + r] = psum[r];
    }
    __syncthreads();
    if (tid < 16) {
        float ssum = 0.0f;
        #pragma unroll
        for (int ww = 0; ww < 8; ++ww) ssum += red[ww][tid];
        rowlse_s[tid] = logf(ssum);
    }
    __syncthreads();

    #pragma unroll
    for (int j = 0; j < 4; ++j) {
        int n = n0 + j * 16 + l16;
        #pragma unroll
        for (int r = 0; r < 8; ++r) {
            int row = 8 * hh + r;
            float lp = s[j][r] - rowmax_s[row] - rowlse_s[row];
            size_t idx = ((size_t)b * T1_SZ + m0 + row) * T2_SZ + n;
            out[idx]       = expf(lp);
            out[TOT + idx] = lp;
        }
    }
}

// ---------------------------------------------------------------------------
static inline size_t align256(size_t x) { return (x + 255) & ~(size_t)255; }

extern "C" void kernel_launch(void* const* d_in, const int* in_sizes, int n_in,
                              void* d_out, int out_size, void* d_ws, size_t ws_size,
                              hipStream_t stream) {
    (void)in_sizes; (void)n_in; (void)out_size; (void)ws_size;
    const float* spec = (const float*)d_in[0];
    const float* text = (const float*)d_in[2];
    const unsigned char* mask = (const unsigned char*)d_in[4];
    const float* qw1 = (const float*)d_in[5];
    const float* qb1 = (const float*)d_in[6];
    const float* qw2 = (const float*)d_in[7];
    const float* qb2 = (const float*)d_in[8];
    const float* qw3 = (const float*)d_in[9];
    const float* qb3 = (const float*)d_in[10];
    const float* kw1 = (const float*)d_in[11];
    const float* kb1 = (const float*)d_in[12];
    const float* kw2 = (const float*)d_in[13];
    const float* kb2 = (const float*)d_in[14];

    char* ws = (char*)d_ws;
    size_t off = 0;
    _Float16* qf = (_Float16*)(ws + off);
    off = align256(off + (size_t)B_SZ * T1_SZ * CATT * 2);
    _Float16* kf = (_Float16*)(ws + off);
    off = align256(off + (size_t)B_SZ * T2_SZ * CATT * 2);
    float* q2 = (float*)(ws + off);
    off = align256(off + (size_t)B_SZ * T1_SZ * 4);
    float* k2 = (float*)(ws + off);
    off = align256(off + (size_t)B_SZ * T2_SZ * 4);
    _Float16* textf = (_Float16*)(ws + off);
    off = align256(off + (size_t)B_SZ * (T2_SZ + 2) * CTXT * 2);
    _Float16* specf = (_Float16*)(ws + off);
    off = align256(off + (size_t)B_SZ * (T1_SZ + 4) * 96 * 2);
    _Float16* pkw1 = (_Float16*)(ws + off); off = align256(off + (size_t)NPK1 * 2);
    _Float16* pkw2 = (_Float16*)(ws + off); off = align256(off + (size_t)NPK2 * 2);
    _Float16* pqw1 = (_Float16*)(ws + off); off = align256(off + (size_t)NPQ1 * 2);
    _Float16* pqw2 = (_Float16*)(ws + off); off = align256(off + (size_t)NPQ2 * 2);
    _Float16* pqw3 = (_Float16*)(ws + off); off = align256(off + (size_t)NPQ3 * 2);

    {
        size_t n = (size_t)B_SZ * (T2_SZ + 2) * CTXT;
        convert_text_kernel<<<dim3((unsigned)((n + 255) / 256)), 256, 0, stream>>>(
            text, textf);
    }
    {
        size_t n = (size_t)B_SZ * (T1_SZ + 4) * 96;
        convert_spec_kernel<<<dim3((unsigned)((n + 255) / 256)), 256, 0, stream>>>(
            spec, specf);
    }
    {
        int n = NPK1 + NPK2 + NPQ1 + NPQ2 + NPQ3;
        pack_weights_kernel<<<dim3((n + 255) / 256), 256, 0, stream>>>(
            kw1, kw2, qw1, qw2, qw3, pkw1, pkw2, pqw1, pqw2, pqw3);
    }
    k_enc_wmma<<<dim3(T2_SZ / 64, B_SZ), 256, 0, stream>>>(
        textf, pkw1, kb1, pkw2, kb2, kf, k2);
    q_enc_wmma<<<dim3(T1_SZ / 64, B_SZ), 320, 0, stream>>>(
        specf, pqw1, qb1, pqw2, qb2, pqw3, qb3, qf, q2);
    attn_kernel<<<dim3(T1_SZ / 16, B_SZ), 256, 0, stream>>>(
        qf, kf, q2, k2, mask, (float*)d_out);
}
